// SE_GCL_4939212390714
// MI455X (gfx1250) — matrix operand order repaired and checked by reading, not compiled, not run on hardware
//
#include <hip/hip_runtime.h>
#include <hip/hip_bf16.h>

typedef __attribute__((ext_vector_type(16))) __bf16 v16bf;
typedef __attribute__((ext_vector_type(8)))  __bf16 v8bf;
typedef __attribute__((ext_vector_type(8)))  float  v8f;

// ---------------- param offsets (jax pytree flatten: sorted dict keys, b<w) ----
// chain_local @0, chain_sub @61824, gedge @115456, gedge_ln @201856,
// gnode @202112, gnode_ln @432128, local_ln @432384, prox_local @432640,
// prox_sub @494464, prox_w @548096, sub_ln @560577  (total 560833)
#define CL_B1 0
#define CL_B2 28800
#define CL_B3 45312
#define CS_B1 61824
#define CS_B2 82432
#define CS_B3 98944
#define GE_B1 115456
#define GE_B2 168832
#define GE_B3 185344
#define GE_LN_B 201856
#define GE_LN_G 201984
#define GN_B1 202112
#define GN_B2 333440
#define GN_B3 399232
#define GN_LN_B 432128
#define GN_LN_G 432256
#define LOC_LN_B 432384
#define LOC_LN_G 432512
#define PL_B1 432640
#define PL_B2 461440
#define PL_B3 477952
#define PS_B1 494464
#define PS_B2 515072
#define PS_B3 531584
#define PW_B1 548096
#define PW_W1 548160
#define PW_B2 556352
#define PW_W2 556416
#define PW_B3 560512
#define PW_W3 560513
#define SUB_LN_B 560577
#define SUB_LN_G 560705

// packed bf16 weight offsets (elements) in workspace
#define WP_PL1 0
#define WP_PL2 28672
#define WP_PL3 45056
#define WP_CL1 61440
#define WP_CL2 90112
#define WP_CL3 106496
#define WP_PS1 122880
#define WP_PS2 143360
#define WP_PS3 159744
#define WP_CS1 176128
#define WP_CS2 196608
#define WP_CS3 212992
#define WP_GN1 229376
#define WP_GN2 360448
#define WP_GN3 425984
#define WP_GE1 458752
#define WP_GE2 512000
#define WP_GE3 528384
#define WP_TOTAL 544768

#define NB 2
#define NN 384
#define KP 30
#define KC 2

__device__ __forceinline__ __bf16 f2bf(float f) {
  union { float f; unsigned u; } x; x.f = f;
  unsigned r = x.u + 0x7FFFu + ((x.u >> 16) & 1u); // round-to-nearest-even
  union { unsigned short s; __bf16 b; } y; y.s = (unsigned short)(r >> 16);
  return y.b;
}

__device__ __forceinline__ float rbf_val(float d, int m) {
  float mu = 20.0f * (float)m / 31.0f;
  float t = (d - mu) * 1.6f; // sigma = 20/32 = 0.625
  return __expf(-t * t);
}

// One 3-layer-MLP layer on a 16-row tile staged in LDS (bf16, row-major,
// stride = din). Weights are pre-packed in WMMA B-fragment lane order.
// 256 threads = 8 waves; wave w covers column-tiles t = w, w+8, ...
__device__ __forceinline__ void wmma_layer(const __bf16* As, int din,
                                           const __bf16* Wp, const float* bias,
                                           int dout, bool relu,
                                           __bf16* OutB, float* OutF) {
  const int tid  = threadIdx.x;
  const int wave = tid >> 5;
  const int lane = tid & 31;
  const int r    = lane & 15;
  const int hi   = lane >> 4;
  const int nkb  = din >> 5;
  const int ntile = dout >> 4;
  for (int t = wave; t < ntile; t += 8) {
    v8f acc;
#pragma unroll
    for (int i = 0; i < 8; ++i) acc[i] = 0.0f;
    for (int kb = 0; kb < nkb; ++kb) {
      // A fragment: lane half hi owns K = hi*8 + [0..7] and 16 + hi*8 + [0..7]
      const __bf16* arow = As + r * din + kb * 32 + hi * 8;
      v8bf a0 = *(const v8bf*)(arow);
      v8bf a1 = *(const v8bf*)(arow + 16);
      v16bf a;
#pragma unroll
      for (int q = 0; q < 8; ++q) { a[q] = a0[q]; a[8 + q] = a1[q]; }
      // B fragment: 16 contiguous bf16 per lane from the packed layout
      v16bf b = *(const v16bf*)(Wp + ((((size_t)t * nkb) + kb) * 32 + lane) * 16);
      acc = __builtin_amdgcn_wmma_f32_16x16x32_bf16(
          false, a, false, b, (short)0, acc, false, false);
    }
    const int col = t * 16 + (lane & 15);
    const float bi = bias[col];
#pragma unroll
    for (int i = 0; i < 8; ++i) {
      int row = hi * 8 + i;           // C layout: VGPR i -> M=i (lo half), M=8+i (hi half)
      float v = acc[i] + bi;
      if (relu) v = fmaxf(v, 0.0f);
      if (OutB) OutB[row * dout + col] = f2bf(v);
      if (OutF) OutF[row * dout + col] = v;
    }
  }
}

// LayerNorm over 128 features, 16 rows, 256 threads (16 lanes per row).
__device__ __forceinline__ void layernorm16(float* S, const float* g, const float* b) {
  const int tid = threadIdx.x;
  const int row = tid >> 4;
  const int sub = tid & 15;
  float s = 0.0f;
  for (int e = sub; e < 128; e += 16) s += S[row * 128 + e];
  for (int o = 8; o > 0; o >>= 1) s += __shfl_xor(s, o, 16);
  float mu = s * (1.0f / 128.0f);
  float v = 0.0f;
  for (int e = sub; e < 128; e += 16) { float d = S[row * 128 + e] - mu; v += d * d; }
  for (int o = 8; o > 0; o >>= 1) v += __shfl_xor(v, o, 16);
  float rinv = rsqrtf(v * (1.0f / 128.0f) + 1e-5f);
  for (int e = sub; e < 128; e += 16)
    S[row * 128 + e] = (S[row * 128 + e] - mu) * rinv * g[e] + b[e];
}

// ------------------------------------------------------------------ pack ----
__global__ __launch_bounds__(256) void pack_weight_kernel(const float* __restrict__ W,
                                                          __bf16* __restrict__ Wp,
                                                          int din, int dout) {
  int p = blockIdx.x * 256 + threadIdx.x;
  if (p >= din * dout) return;
  int nkb = din >> 5;
  int e  = p & 15;
  int l  = (p >> 4) & 31;
  int kb = (p >> 9) % nkb;
  int t  = p / (nkb << 9);
  int K  = kb * 32 + ((l >= 16) ? 8 : 0) + (e & 7) + ((e >> 3) << 4);
  int c  = t * 16 + (l & 15);
  Wp[p] = f2bf(W[(size_t)K * dout + c]);
}

// ------------------------------------------- fused local+sub branch kernel ----
__global__ __launch_bounds__(256) void local_branch_kernel(
    const int* __restrict__ graph, const float* __restrict__ dists,
    const float* __restrict__ cmask, const float* __restrict__ edge_embed,
    const float* __restrict__ atom_embed, const float* __restrict__ node_mask,
    const __bf16* __restrict__ WP, const float* __restrict__ P, int K,
    int wloc, int wsub, int bl1, int bl2, int bl3, int bs1, int bs2, int bs3,
    float* __restrict__ wsn) {
  __shared__ __align__(16) __bf16 s_in[16 * 224];
  __shared__ __align__(16) __bf16 s_h[16 * 128];
  __shared__ __align__(16) __bf16 s_h2[16 * 128];
  __shared__ __align__(16) float  s_f[16 * 128];
  __shared__ float s_ln[4 * 128];
  __shared__ float s_sn[128];
  __shared__ float s_w1[64];
  __shared__ float s_w2[64];
  __shared__ float s_wout;

  const int idx = blockIdx.x;
  const int k = idx % K;
  const int n = (idx / K) % NN;
  const int b = idx / (K * NN);
  const int tid = threadIdx.x;
  const int nb = graph[(b * NN + n) * K + k];

  // init tile: rows r = ai*4+aj; cols [ee(128) | atom_ai(32) | atom_aj(32) | rbf(32)]
  for (int p = tid; p < 16 * 224; p += 256) {
    int r = p / 224, c = p % 224;
    int ai = r >> 2, aj = r & 3;
    float v;
    if (c < 128)       v = edge_embed[(((size_t)b * NN + n) * NN + nb) * 128 + c];
    else if (c < 160)  v = atom_embed[ai * 32 + (c - 128)];
    else if (c < 192)  v = atom_embed[aj * 32 + (c - 160)];
    else {
      float d = dists[((((size_t)b * NN + n) * 4 + ai) * K + k) * 4 + aj];
      v = rbf_val(d, c - 192);
    }
    s_in[p] = f2bf(v);
  }
  __syncthreads();
  wmma_layer(s_in, 224, WP + wloc, P + bl1, 128, true, s_h, nullptr);
  __syncthreads();
  wmma_layer(s_h, 128, WP + wloc + 224 * 128, P + bl2, 128, true, s_h2, nullptr);
  __syncthreads();
  wmma_layer(s_h2, 128, WP + wloc + 224 * 128 + 128 * 128, P + bl3, 128, false, nullptr, s_f);
  __syncthreads();

  float m = node_mask[b * NN + n];
  if (cmask) m *= cmask[(b * NN + n) * K + k];
  { int row = tid >> 4;
    for (int c = tid & 15; c < 128; c += 16) s_f[row * 128 + c] *= m; }
  layernorm16(s_f, P + LOC_LN_G, P + LOC_LN_B);
  __syncthreads();

  // mean over aj -> ln_node[ai][128]
  for (int p = tid; p < 4 * 128; p += 256) {
    int ai = p >> 7, c = p & 127;
    float s = 0.0f;
    for (int aj = 0; aj < 4; ++aj) s += s_f[(ai * 4 + aj) * 128 + c];
    s_ln[p] = 0.25f * s;
  }
  __syncthreads();

  // sub input tile 16x160 (rows 0..3 valid, rest zero-padded)
  for (int p = tid; p < 16 * 160; p += 256) {
    int r = p / 160, c = p % 160;
    float v = 0.0f;
    if (r < 4) v = (c < 32) ? atom_embed[r * 32 + c] : s_ln[r * 128 + (c - 32)];
    s_in[p] = f2bf(v);
  }
  __syncthreads();
  wmma_layer(s_in, 160, WP + wsub, P + bs1, 128, true, s_h, nullptr);
  __syncthreads();
  wmma_layer(s_h, 128, WP + wsub + 160 * 128, P + bs2, 128, true, s_h2, nullptr);
  __syncthreads();
  wmma_layer(s_h2, 128, WP + wsub + 160 * 128 + 128 * 128, P + bs3, 128, false, nullptr, s_f);
  __syncthreads();
  { int row = tid >> 4;
    for (int c = tid & 15; c < 128; c += 16) s_f[row * 128 + c] *= m; }
  layernorm16(s_f, P + SUB_LN_G, P + SUB_LN_B);
  __syncthreads();

  // sn = mean over ai (rows 0..3)
  for (int c = tid; c < 128; c += 256) {
    float s = 0.0f;
    for (int ai = 0; ai < 4; ++ai) s += s_f[ai * 128 + c];
    s_sn[c] = 0.25f * s;
  }
  __syncthreads();

  // attention weight MLP 128->64->64->1 (always prox_w params, per reference)
  if (tid < 64) {
    float s = P[PW_B1 + tid];
    for (int c = 0; c < 128; ++c) s += s_sn[c] * P[PW_W1 + c * 64 + tid];
    s_w1[tid] = fmaxf(s, 0.0f);
  }
  __syncthreads();
  if (tid < 64) {
    float s = P[PW_B2 + tid];
    for (int c = 0; c < 64; ++c) s += s_w1[c] * P[PW_W2 + c * 64 + tid];
    s_w2[tid] = fmaxf(s, 0.0f);
  }
  __syncthreads();
  if (tid == 0) {
    float s = P[PW_B3];
    for (int c = 0; c < 64; ++c) s += s_w2[c] * P[PW_W3 + c];
    s_wout = s;
  }
  __syncthreads();
  float w = s_wout;
  for (int c = tid; c < 128; c += 256)
    wsn[(size_t)idx * 128 + c] = w * s_sn[c];
}

// ---------------------------------------------------------------- reduce ----
__global__ __launch_bounds__(128) void reduce_mean_k(const float* __restrict__ wsn,
                                                     float* __restrict__ out, int K) {
  int bn = blockIdx.x, c = threadIdx.x;
  float s = 0.0f;
  for (int k = 0; k < K; ++k) s += wsn[((size_t)bn * K + k) * 128 + c];
  out[(size_t)bn * 128 + c] = s / (float)K;
}

// ----------------------------------------------------------------- gnode ----
__global__ __launch_bounds__(256) void gnode_kernel(
    const float* __restrict__ prox_node, const float* __restrict__ chain_node,
    const float* __restrict__ node_embed, const float* __restrict__ node_mask,
    const __bf16* __restrict__ WP, const float* __restrict__ P,
    float* __restrict__ gn_out) {
  __shared__ __align__(16) __bf16 s_in[16 * 512];
  __shared__ __align__(16) __bf16 s_h[16 * 256];
  __shared__ __align__(16) __bf16 s_h2[16 * 256];
  __shared__ __align__(16) float  s_f[16 * 128];
  const int tid = threadIdx.x;
  const int r0 = blockIdx.x * 16;
  for (int idx = tid; idx < 16 * 512; idx += 256) {
    int r = idx >> 9, c = idx & 511;
    int bn = r0 + r;
    float v;
    if (c < 128)      v = prox_node[(size_t)bn * 128 + c];
    else if (c < 256) v = chain_node[(size_t)bn * 128 + (c - 128)];
    else              v = node_embed[(size_t)bn * 256 + (c - 256)];
    s_in[idx] = f2bf(v);
  }
  __syncthreads();
  wmma_layer(s_in, 512, WP + WP_GN1, P + GN_B1, 256, true, s_h, nullptr);
  __syncthreads();
  wmma_layer(s_h, 256, WP + WP_GN2, P + GN_B2, 256, true, s_h2, nullptr);
  __syncthreads();
  wmma_layer(s_h2, 256, WP + WP_GN3, P + GN_B3, 128, false, nullptr, s_f);
  __syncthreads();
  { int row = tid >> 4; float m = node_mask[r0 + row];
    for (int c = tid & 15; c < 128; c += 16) s_f[row * 128 + c] *= m; }
  layernorm16(s_f, P + GN_LN_G, P + GN_LN_B);
  __syncthreads();
  for (int idx = tid; idx < 16 * 128; idx += 256) {
    int r = idx >> 7, c = idx & 127;
    gn_out[((size_t)(r0 + r)) * 128 + c] = s_f[idx];
  }
}

// ----------------------------------------------------------------- gedge ----
__global__ __launch_bounds__(256) void gedge_kernel(
    const float* __restrict__ gn, const float* __restrict__ edge_embed,
    const float* __restrict__ ca_d, const float* __restrict__ edge_mask,
    const __bf16* __restrict__ WP, const float* __restrict__ P,
    float* __restrict__ ge_out) {
  __shared__ __align__(16) __bf16 s_in[16 * 416];
  __shared__ __align__(16) __bf16 s_h[16 * 128];
  __shared__ __align__(16) __bf16 s_h2[16 * 128];
  __shared__ __align__(16) float  s_f[16 * 128];
  const int blk = blockIdx.x;
  const int jt = blk % (NN / 16);
  const int i  = (blk / (NN / 16)) % NN;
  const int bb = blk / ((NN / 16) * NN);
  const int tid = threadIdx.x;
  // input: [gn[b,j](128) | gn[b,i](128) | edge_embed[b,i,j](128) | rbf(ca_d)(32)]
  for (int idx = tid; idx < 16 * 416; idx += 256) {
    int r = idx / 416, c = idx % 416;
    int j = jt * 16 + r;
    float v;
    if (c < 128)      v = gn[((size_t)bb * NN + j) * 128 + c];
    else if (c < 256) v = gn[((size_t)bb * NN + i) * 128 + (c - 128)];
    else if (c < 384) v = edge_embed[(((size_t)bb * NN + i) * NN + j) * 128 + (c - 256)];
    else              v = rbf_val(ca_d[((size_t)bb * NN + i) * NN + j], c - 384);
    s_in[idx] = f2bf(v);
  }
  __syncthreads();
  wmma_layer(s_in, 416, WP + WP_GE1, P + GE_B1, 128, true, s_h, nullptr);
  __syncthreads();
  wmma_layer(s_h, 128, WP + WP_GE2, P + GE_B2, 128, true, s_h2, nullptr);
  __syncthreads();
  wmma_layer(s_h2, 128, WP + WP_GE3, P + GE_B3, 128, false, nullptr, s_f);
  __syncthreads();
  { int row = tid >> 4; int j = jt * 16 + row;
    float m = edge_mask[((size_t)bb * NN + i) * NN + j];
    for (int c = tid & 15; c < 128; c += 16) s_f[row * 128 + c] *= m; }
  layernorm16(s_f, P + GE_LN_G, P + GE_LN_B);
  __syncthreads();
  for (int idx = tid; idx < 16 * 128; idx += 256) {
    int r = idx >> 7, c = idx & 127;
    ge_out[((((size_t)bb * NN + i) * NN) + jt * 16 + r) * 128 + c] = s_f[idx];
  }
}

// ------------------------------------------------------------------ host ----
extern "C" void kernel_launch(void* const* d_in, const int* in_sizes, int n_in,
                              void* d_out, int out_size, void* d_ws, size_t ws_size,
                              hipStream_t stream) {
  const float* P          = (const float*)d_in[0];
  const float* node_embed = (const float*)d_in[1];
  const float* edge_embed = (const float*)d_in[2];
  const float* atom_embed = (const float*)d_in[3];
  const int*   prox_g     = (const int*)d_in[4];
  const float* prox_d     = (const float*)d_in[5];
  const int*   chain_g    = (const int*)d_in[6];
  const float* chain_d    = (const float*)d_in[7];
  const float* chain_m    = (const float*)d_in[8];
  const float* ca_d       = (const float*)d_in[9];
  const float* node_mask  = (const float*)d_in[10];
  const float* edge_mask  = (const float*)d_in[11];

  float* out    = (float*)d_out;
  float* gn_out = out;
  float* ge_out = out + (size_t)NB * NN * 128;

  char* ws = (char*)d_ws;
  __bf16* WP        = (__bf16*)ws;                         // 1,089,536 B
  float*  wsn_p     = (float*)(ws + 1089536);              // 11,796,480 B
  float*  wsn_c     = (float*)(ws + 12886016);             //    786,432 B
  float*  prox_node = (float*)(ws + 13672448);             //    393,216 B
  float*  chain_node= (float*)(ws + 14065664);             //    393,216 B

  struct Pack { int p_off, wp_off, din, dout; };
  static const Pack packs[18] = {
    {432768, WP_PL1, 224, 128}, {461568, WP_PL2, 128, 128}, {478080, WP_PL3, 128, 128},
    {128,    WP_CL1, 224, 128}, {28928,  WP_CL2, 128, 128}, {45440,  WP_CL3, 128, 128},
    {494592, WP_PS1, 160, 128}, {515200, WP_PS2, 128, 128}, {531712, WP_PS3, 128, 128},
    {61952,  WP_CS1, 160, 128}, {82560,  WP_CS2, 128, 128}, {99072,  WP_CS3, 128, 128},
    {202368, WP_GN1, 512, 256}, {333696, WP_GN2, 256, 256}, {399360, WP_GN3, 256, 128},
    {115584, WP_GE1, 416, 128}, {168960, WP_GE2, 128, 128}, {185472, WP_GE3, 128, 128},
  };
  for (int m = 0; m < 18; ++m) {
    int total = packs[m].din * packs[m].dout;
    pack_weight_kernel<<<(total + 255) / 256, 256, 0, stream>>>(
        P + packs[m].p_off, WP + packs[m].wp_off, packs[m].din, packs[m].dout);
  }

  local_branch_kernel<<<NB * NN * KP, 256, 0, stream>>>(
      prox_g, prox_d, nullptr, edge_embed, atom_embed, node_mask, WP, P, KP,
      WP_PL1, WP_PS1, PL_B1, PL_B2, PL_B3, PS_B1, PS_B2, PS_B3, wsn_p);

  local_branch_kernel<<<NB * NN * KC, 256, 0, stream>>>(
      chain_g, chain_d, chain_m, edge_embed, atom_embed, node_mask, WP, P, KC,
      WP_CL1, WP_CS1, CL_B1, CL_B2, CL_B3, CS_B1, CS_B2, CS_B3, wsn_c);

  reduce_mean_k<<<NB * NN, 128, 0, stream>>>(wsn_p, prox_node, KP);
  reduce_mean_k<<<NB * NN, 128, 0, stream>>>(wsn_c, chain_node, KC);

  gnode_kernel<<<(NB * NN) / 16, 256, 0, stream>>>(
      prox_node, chain_node, node_embed, node_mask, WP, P, gn_out);

  gedge_kernel<<<NB * NN * (NN / 16), 256, 0, stream>>>(
      gn_out, edge_embed, ca_d, edge_mask, WP, P, ge_out);
}
